// LookupFFN_68719476736163
// MI455X (gfx1250) — matrix operand order repaired and checked by reading, not compiled, not run on hardware
//
#include <hip/hip_runtime.h>
#include <hip/hip_bf16.h>
#include <stdint.h>

#define N_ROWS 16384
#define D_IN   1024
#define NBKT   1024
#define D_OUT  1024

typedef __bf16 v16bf __attribute__((ext_vector_type(16)));
typedef float  v8f   __attribute__((ext_vector_type(8)));

union Frag {
    v16bf v;
    unsigned short h[16];
    uint4 q[2];
};

__device__ __forceinline__ unsigned short f2bf_rne(float f) {
    uint32_t u = __builtin_bit_cast(uint32_t, f);
    u += 0x7FFFu + ((u >> 16) & 1u);
    return (unsigned short)(u >> 16);
}
__device__ __forceinline__ float bf2f(unsigned short h) {
    return __builtin_bit_cast(float, (uint32_t)h << 16);
}

__device__ __forceinline__ void cvt16(const float* v, Frag& hi, Frag& lo) {
#pragma unroll
    for (int i = 0; i < 16; ++i) {
        unsigned short hb = f2bf_rne(v[i]);
        hi.h[i] = hb;
        lo.h[i] = f2bf_rne(v[i] - bf2f(hb));
    }
}

// Split f32 -> (bf16 hi, bf16 lo) planes; 8 elements per thread.
__global__ void cvt_hilo_kernel(const float* __restrict__ in,
                                unsigned short* __restrict__ hi,
                                unsigned short* __restrict__ lo, int n8) {
    int i = blockIdx.x * blockDim.x + threadIdx.x;
    if (i >= n8) return;
    const float4* p = (const float4*)in + (size_t)i * 2;
    float4 a = p[0], b = p[1];
    float v[8] = {a.x, a.y, a.z, a.w, b.x, b.y, b.z, b.w};
    union { uint4 q; unsigned short h[8]; } H, Lo;
#pragma unroll
    for (int j = 0; j < 8; ++j) {
        unsigned short hb = f2bf_rne(v[j]);
        H.h[j]  = hb;
        Lo.h[j] = f2bf_rne(v[j] - bf2f(hb));
    }
    *(uint4*)(hi + (size_t)i * 8) = H.q;
    *(uint4*)(lo + (size_t)i * 8) = Lo.q;
}

// One wave32 owns 16 rows of x. Buckets in 8 groups of 128 (8 WMMA tiles),
// K=1024 in chunks of 32. 3-term bf16 split GEMM into f32 accumulators,
// issued term-major so consecutive WMMAs hit different accumulators.
template <bool PRE_X, bool PRE_R>
__global__ __launch_bounds__(256) void lookup_ffn_kernel(
    const float* __restrict__ x, const float* __restrict__ R,
    const float* __restrict__ L,
    const unsigned short* __restrict__ xh, const unsigned short* __restrict__ xl,
    const unsigned short* __restrict__ rh, const unsigned short* __restrict__ rl,
    float* __restrict__ out) {
    const int lane = threadIdx.x & 31;
    const int wave = threadIdx.x >> 5;
    const int rb   = blockIdx.x * 8 + wave;   // row-block id [0,1024)
    const int m0   = rb * 16;
    const int mr   = lane & 15;               // A row / B,C column within tile
    const int c    = lane >> 4;               // half-wave selector
    const int m    = m0 + mr;

    // Hoisted per-lane row bases.
    const unsigned short* xrh_base = PRE_X ? xh + (size_t)m * D_IN + 8 * c : nullptr;
    const unsigned short* xrl_base = PRE_X ? xl + (size_t)m * D_IN + 8 * c : nullptr;
    const float*          xf_base  = x + (size_t)m * D_IN + 8 * c;

    float bestv[8];
    int   bestc[8];                           // tile code: g*8 + t
#pragma unroll
    for (int i = 0; i < 8; ++i) { bestv[i] = -__builtin_inff(); bestc[i] = 0; }

#pragma unroll 1
    for (int g = 0; g < 8; ++g) {             // 8 bucket groups x 128 buckets
        // Per-lane B row bases for this group (tile stride folds to imm offset).
        const size_t brow = (size_t)(g * 128 + mr) * D_IN + 8 * c;
        const unsigned short* rh_g = PRE_R ? rh + brow : nullptr;
        const unsigned short* rl_g = PRE_R ? rl + brow : nullptr;
        const float*          rf_g = R + brow;

        v8f acc[8];
#pragma unroll
        for (int t = 0; t < 8; ++t)
#pragma unroll
            for (int i = 0; i < 8; ++i) acc[t][i] = 0.f;

#pragma unroll 1
        for (int kc = 0; kc < 32; ++kc) {     // K chunks of 32
            const int K0 = kc * 32;
            Frag ah, al;
            if (PRE_X) {
                ah.q[0] = *(const uint4*)(xrh_base + K0);
                ah.q[1] = *(const uint4*)(xrh_base + K0 + 16);
                al.q[0] = *(const uint4*)(xrl_base + K0);
                al.q[1] = *(const uint4*)(xrl_base + K0 + 16);
            } else {
                const float* xr = xf_base + K0;
                float4 f0 = *(const float4*)(xr);
                float4 f1 = *(const float4*)(xr + 4);
                float4 f2 = *(const float4*)(xr + 16);
                float4 f3 = *(const float4*)(xr + 20);
                float v[16] = {f0.x,f0.y,f0.z,f0.w, f1.x,f1.y,f1.z,f1.w,
                               f2.x,f2.y,f2.z,f2.w, f3.x,f3.y,f3.z,f3.w};
                cvt16(v, ah, al);
            }

            Frag bh[8];
#pragma unroll
            for (int t = 0; t < 8; ++t) {
                if (PRE_R) {
                    const unsigned short* p = rh_g + (size_t)t * 16 * D_IN + K0;
                    bh[t].q[0] = *(const uint4*)(p);
                    bh[t].q[1] = *(const uint4*)(p + 16);
                } else {
                    const float* rr = rf_g + (size_t)t * 16 * D_IN + K0;
                    float4 f0 = *(const float4*)(rr);
                    float4 f1 = *(const float4*)(rr + 4);
                    float4 f2 = *(const float4*)(rr + 16);
                    float4 f3 = *(const float4*)(rr + 20);
                    float v[16] = {f0.x,f0.y,f0.z,f0.w, f1.x,f1.y,f1.z,f1.w,
                                   f2.x,f2.y,f2.z,f2.w, f3.x,f3.y,f3.z,f3.w};
                    Frag dummy;
                    cvt16(v, bh[t], dummy);   // hi only here; lo below
                }
            }
            // Term 1: A_hi * B_hi  (8 independent accumulators back-to-back)
#pragma unroll
            for (int t = 0; t < 8; ++t)
                acc[t] = __builtin_amdgcn_wmma_f32_16x16x32_bf16(
                    false, ah.v, false, bh[t].v, (short)0, acc[t], false, false);
            // Term 2: A_lo * B_hi
#pragma unroll
            for (int t = 0; t < 8; ++t)
                acc[t] = __builtin_amdgcn_wmma_f32_16x16x32_bf16(
                    false, al.v, false, bh[t].v, (short)0, acc[t], false, false);

            Frag bl[8];
#pragma unroll
            for (int t = 0; t < 8; ++t) {
                if (PRE_R) {
                    const unsigned short* p = rl_g + (size_t)t * 16 * D_IN + K0;
                    bl[t].q[0] = *(const uint4*)(p);
                    bl[t].q[1] = *(const uint4*)(p + 16);
                } else {
                    const float* rr = rf_g + (size_t)t * 16 * D_IN + K0;
                    float4 f0 = *(const float4*)(rr);
                    float4 f1 = *(const float4*)(rr + 4);
                    float4 f2 = *(const float4*)(rr + 16);
                    float4 f3 = *(const float4*)(rr + 20);
                    float v[16] = {f0.x,f0.y,f0.z,f0.w, f1.x,f1.y,f1.z,f1.w,
                                   f2.x,f2.y,f2.z,f2.w, f3.x,f3.y,f3.z,f3.w};
                    Frag dummy;
                    cvt16(v, dummy, bl[t]);   // lo only
                }
            }
            // Term 3: A_hi * B_lo
#pragma unroll
            for (int t = 0; t < 8; ++t)
                acc[t] = __builtin_amdgcn_wmma_f32_16x16x32_bf16(
                    false, ah.v, false, bl[t].v, (short)0, acc[t], false, false);
        }

        // Lane-local fold: per lane the bucket index (code*16 + mr) is strictly
        // increasing over (g, t), so strict '>' == first-max semantics.
#pragma unroll
        for (int t = 0; t < 8; ++t) {
#pragma unroll
            for (int e = 0; e < 8; ++e) {
                float val = acc[t][e];
                if (val > bestv[e]) { bestv[e] = val; bestc[e] = g * 8 + t; }
            }
        }
    }

    // Final cross-lane argmax per row element (within each 16-lane half;
    // xor masks < 16 never cross the half boundary). Min-index tie-break.
    int rowIdx[8];
#pragma unroll
    for (int e = 0; e < 8; ++e) {
        float val = bestv[e];
        int   idx = bestc[e] * 16 + mr;       // full bucket index
#pragma unroll
        for (int off = 1; off < 16; off <<= 1) {
            float ov = __shfl_xor(val, off);
            int   oi = __shfl_xor(idx, off);
            if (ov > val || (ov == val && oi < idx)) { val = ov; idx = oi; }
        }
        rowIdx[e] = idx;                      // identical across the 16-lane half
    }

    // Gather: lanes 0-15 hold rows 0-7, lanes 16-31 hold rows 8-15.
#pragma unroll
    for (int mloc = 0; mloc < 16; ++mloc) {
        int idx = __shfl(rowIdx[mloc & 7], (mloc >> 3) << 4);
        const float4* src = (const float4*)(L + (size_t)idx * D_OUT);
        float4*       dst = (float4*)(out + (size_t)(m0 + mloc) * D_OUT);
        for (int j = lane; j < D_OUT / 4; j += 32) dst[j] = src[j];
    }
}

extern "C" void kernel_launch(void* const* d_in, const int* in_sizes, int n_in,
                              void* d_out, int out_size, void* d_ws, size_t ws_size,
                              hipStream_t stream) {
    const float* x = (const float*)d_in[0];
    const float* R = (const float*)d_in[1];
    const float* L = (const float*)d_in[2];
    float* out = (float*)d_out;

    const size_t xHalf = (size_t)N_ROWS * D_IN * sizeof(unsigned short); // 32 MB
    const size_t rHalf = (size_t)NBKT   * D_IN * sizeof(unsigned short); //  2 MB
    unsigned char* ws = (unsigned char*)d_ws;

    const int nx8 = N_ROWS * D_IN / 8;
    const int nr8 = NBKT   * D_IN / 8;
    dim3 mainGrid(N_ROWS / 16 / 8);   // 128 blocks x 8 waves = 1024 row-blocks
    dim3 mainBlk(256);

    if (ws_size >= 2 * xHalf + 2 * rHalf) {
        unsigned short* xh = (unsigned short*)(ws);
        unsigned short* xl = (unsigned short*)(ws + xHalf);
        unsigned short* rh = (unsigned short*)(ws + 2 * xHalf);
        unsigned short* rl = (unsigned short*)(ws + 2 * xHalf + rHalf);
        cvt_hilo_kernel<<<(nx8 + 255) / 256, 256, 0, stream>>>(x, xh, xl, nx8);
        cvt_hilo_kernel<<<(nr8 + 255) / 256, 256, 0, stream>>>(R, rh, rl, nr8);
        lookup_ffn_kernel<true, true><<<mainGrid, mainBlk, 0, stream>>>(
            x, R, L, xh, xl, rh, rl, out);
    } else if (ws_size >= 2 * rHalf) {
        unsigned short* rh = (unsigned short*)(ws);
        unsigned short* rl = (unsigned short*)(ws + rHalf);
        cvt_hilo_kernel<<<(nr8 + 255) / 256, 256, 0, stream>>>(R, rh, rl, nr8);
        lookup_ffn_kernel<false, true><<<mainGrid, mainBlk, 0, stream>>>(
            x, R, L, nullptr, nullptr, rh, rl, out);
    } else {
        lookup_ffn_kernel<false, false><<<mainGrid, mainBlk, 0, stream>>>(
            x, R, L, nullptr, nullptr, nullptr, nullptr, out);
    }
}
